// Attention_21242908246407
// MI455X (gfx1250) — matrix operand order repaired and checked by reading, not compiled, not run on hardware
//
#include <hip/hip_runtime.h>
#include <hip/hip_bf16.h>

typedef __attribute__((ext_vector_type(16))) _Float16 v16h;
typedef __attribute__((ext_vector_type(8)))  float    v8f;
typedef __attribute__((ext_vector_type(4)))  unsigned int u32x4;
typedef __attribute__((ext_vector_type(8)))  int i32x8;
typedef __attribute__((ext_vector_type(4)))  int i32x4;

#define Bsz 64
#define Ssz 4096
#define Dsz 256
#define LDSK 264              // padded LDS row stride (f16 elems): 528B -> lanes land on distinct banks
#define BLK_ROWS 256          // x rows per block
#define ROWS_PER_WAVE 32      // 2 WMMA row-sets per wave -> 2 MACs per B-fragment load

union FragU { uint4 q[2]; v16h v; };

__device__ __forceinline__ float fast_tanh(float v) {
#if __has_builtin(__builtin_amdgcn_tanhf)
    return __builtin_amdgcn_tanhf(v);        // v_tanh_f32 (gfx1250 TRANS op)
#else
    float c = fminf(fmaxf(v, -9.0f), 9.0f);  // branchless fallback
    float e = __expf(2.0f * c);
    return (e - 1.0f) / (e + 1.0f);
#endif
}

// ---------------- kernel 0: W[k][n] (f32) -> Wt[n][k] (f16) in workspace ----------------
__global__ __launch_bounds__(256) void
prep_wt_kernel(const float* __restrict__ W, _Float16* __restrict__ Wt) {
    int idx = blockIdx.x * 256 + threadIdx.x;       // 0..65535
    int n = idx >> 8, k = idx & 255;
    Wt[n * Dsz + k] = (_Float16)W[k * Dsz + n];
}

// ---------------- kernel 1: scores[b,s] = u . tanh(x W + b), WMMA f16 ----------------
// 270 KB LDS -> exactly 1 block / WGP (2 waves / SIMD32); min 1 wave/EU lets the
// 128 A-fragment VGPRs stay resident (verified: no scratch spills in round 2).
__global__ __launch_bounds__(256, 1) void
score_kernel(const float* __restrict__ x,
             const _Float16* __restrict__ Wt,
             const float* __restrict__ bias,
             const float* __restrict__ u,
             float* __restrict__ scores) {
    extern __shared__ _Float16 smem[];
    _Float16* w_lds = smem;                    // [256][LDSK], at dynamic-LDS offset 0
    _Float16* x_lds = smem + Dsz * LDSK;       // [BLK_ROWS][LDSK]

    const int tid  = threadIdx.x;
    const int wave = tid >> 5;
    const int lane = tid & 31;
    const int half = lane >> 4;
    const int ln   = lane & 15;
    const size_t rowBase = (size_t)blockIdx.x * BLK_ROWS;

#if __has_builtin(__builtin_amdgcn_tensor_load_to_lds)
    // --- Tensor Data Mover: stage Wt (256x256 f16) global -> LDS with row padding.
    // D# per CDNA5 ISA ch.8: 2D tensor, data_size=2B, tile = whole tensor,
    // pad_interval=128 DWORDs (one 512B row), pad_amount=4 DWORDs (16B) -> 528B rows = LDSK.
    if (wave == 0) {
        unsigned long long ga = (unsigned long long)(uintptr_t)Wt;
        u32x4 g0;
        g0[0] = 1u;                                        // count=1 (valid user descriptor)
        g0[1] = 0u;                                        // lds_addr = 0 (w_lds base)
        g0[2] = (unsigned)(ga & 0xFFFFFFFFu);              // global_addr[31:0]
        g0[3] = (unsigned)(((ga >> 32) & 0x1FFFFFFu) | (2u << 30)); // global_addr[56:32] | type=2
        i32x8 g1;
        g1[0] = (int)((1u << 16)      // data_size = 1 -> 2 bytes
                    | (1u << 20)      // pad_enable
                    | (6u << 22)      // pad_interval code 6 -> 128 DWORDs
                    | (3u << 25));    // pad_amount  code 3 -> 4 DWORDs
        g1[1] = (int)(256u << 16);    // tensor_dim0 = 256 (bits 79:48)
        g1[2] = (int)(256u << 16);    // tensor_dim1 = 256 (bits 111:80)
        g1[3] = (int)(256u << 16);    // tile_dim0   = 256 (bits 127:112)
        g1[4] = (int)256;             // tile_dim1 = 256, tile_dim2 = 0
        g1[5] = (int)256;             // tensor_dim0_stride = 256 elems
        g1[6] = 0;
        g1[7] = 0;
        i32x4 gz4; gz4[0] = 0; gz4[1] = 0; gz4[2] = 0; gz4[3] = 0;
        i32x8 gz8; gz8[0] = 0; gz8[1] = 0; gz8[2] = 0; gz8[3] = 0;
                   gz8[4] = 0; gz8[5] = 0; gz8[6] = 0; gz8[7] = 0;
        // 6-arg form (amdgpu-toolchain / clang-23): groups 2/3 + extra group unused (2-D tensor)
        __builtin_amdgcn_tensor_load_to_lds(g0, g1, gz4, gz4, gz8, 0);
    }
#else
    {
        const uint4* wt4 = (const uint4*)Wt;
        for (int c = tid; c < Dsz * Dsz / 8; c += 256) {
            int n = c >> 5, j = c & 31;
            *(uint4*)(w_lds + n * LDSK + j * 8) = wt4[c];
        }
    }
#endif

    // stage x tile (f32 global, coalesced float4) -> f16 LDS (overlaps the TDM)
    {
        const float4* x4 = (const float4*)x;
        for (int c = tid; c < BLK_ROWS * (Dsz / 4); c += 256) {
            int row = c >> 6, j = c & 63;
            float4 v = x4[(rowBase + row) * (Dsz / 4) + j];
            union { _Float16 h[4]; uint2 u2; } p;
            p.h[0] = (_Float16)v.x; p.h[1] = (_Float16)v.y;
            p.h[2] = (_Float16)v.z; p.h[3] = (_Float16)v.w;
            *(uint2*)(x_lds + row * LDSK + j * 4) = p.u2;
        }
    }
#if __has_builtin(__builtin_amdgcn_tensor_load_to_lds)
    if (wave == 0) __builtin_amdgcn_s_wait_tensorcnt(0);
#endif
    __syncthreads();

    // A fragments (16x32 f16 layout: a[0..7]=K(8*half+i), a[8..15]=K(16+8*half+i)), 2 row-sets
    const _Float16* a0 = x_lds + (wave * ROWS_PER_WAVE + ln) * LDSK;
    const _Float16* a1 = a0 + 16 * LDSK;
    v16h A0[8], A1[8];
#pragma unroll
    for (int kt = 0; kt < 8; ++kt) {
        int k0 = kt * 32;
        FragU f0, f1;
        f0.q[0] = *(const uint4*)(a0 + k0 + 8 * half);
        f0.q[1] = *(const uint4*)(a0 + k0 + 16 + 8 * half);
        f1.q[0] = *(const uint4*)(a1 + k0 + 8 * half);
        f1.q[1] = *(const uint4*)(a1 + k0 + 16 + 8 * half);
        A0[kt] = f0.v; A1[kt] = f1.v;
    }

    float sp0[8] = {0,0,0,0,0,0,0,0};
    float sp1[8] = {0,0,0,0,0,0,0,0};

    for (int nt = 0; nt < 16; ++nt) {
        const int n0 = nt * 16;
        v8f acc0 = {}; v8f acc1 = {};
        const _Float16* brow = w_lds + (n0 + ln) * LDSK + 8 * half;
        // software-pipelined B fragments (prefetch distance 2)
        FragU fb[8];
#pragma unroll
        for (int kt = 0; kt < 2; ++kt) {
            fb[kt].q[0] = *(const uint4*)(brow + kt * 32);
            fb[kt].q[1] = *(const uint4*)(brow + kt * 32 + 16);
        }
#pragma unroll
        for (int kt = 0; kt < 8; ++kt) {
            if (kt + 2 < 8) {
                fb[kt + 2].q[0] = *(const uint4*)(brow + (kt + 2) * 32);
                fb[kt + 2].q[1] = *(const uint4*)(brow + (kt + 2) * 32 + 16);
            }
            acc0 = __builtin_amdgcn_wmma_f32_16x16x32_f16(false, A0[kt], false, fb[kt].v,
                                                          (short)0, acc0, false, false);
            acc1 = __builtin_amdgcn_wmma_f32_16x16x32_f16(false, A1[kt], false, fb[kt].v,
                                                          (short)0, acc1, false, false);
        }
        const float bn = bias[n0 + ln];
        const float un = u[n0 + ln];
#pragma unroll
        for (int r = 0; r < 8; ++r) {                  // acc[r] = D[m=r+8*half][n=ln]
            sp0[r] += fast_tanh(acc0[r] + bn) * un;
            sp1[r] += fast_tanh(acc1[r] + bn) * un;
        }
    }

    // reduce over the 16 n-lanes within each half
#pragma unroll
    for (int off = 8; off >= 1; off >>= 1) {
#pragma unroll
        for (int r = 0; r < 8; ++r) {
            sp0[r] += __shfl_xor(sp0[r], off, 16);
            sp1[r] += __shfl_xor(sp1[r], off, 16);
        }
    }
    if (ln == 0) {
        size_t base = rowBase + (size_t)wave * ROWS_PER_WAVE + 8 * half;
#pragma unroll
        for (int r = 0; r < 8; ++r) {
            scores[base + r]      = sp0[r];
            scores[base + 16 + r] = sp1[r];
        }
    }
}

// ---------------- kernel 2: in-place softmax over S per batch ----------------
__global__ __launch_bounds__(256) void
softmax_kernel(float* __restrict__ w) {
    __shared__ float red[256];
    const int b = blockIdx.x, tid = threadIdx.x;
    float* p = w + (size_t)b * Ssz;
    float loc[16];
    float mx = -3.402823466e38f;
#pragma unroll
    for (int i = 0; i < 16; ++i) { float s = p[tid + i * 256]; loc[i] = s; mx = fmaxf(mx, s); }
    red[tid] = mx; __syncthreads();
    for (int off = 128; off >= 1; off >>= 1) {
        if (tid < off) red[tid] = fmaxf(red[tid], red[tid + off]);
        __syncthreads();
    }
    mx = red[0]; __syncthreads();
    float sum = 0.f;
#pragma unroll
    for (int i = 0; i < 16; ++i) { float e = __expf(loc[i] - mx); loc[i] = e; sum += e; }
    red[tid] = sum; __syncthreads();
    for (int off = 128; off >= 1; off >>= 1) {
        if (tid < off) red[tid] += red[tid + off];
        __syncthreads();
    }
    float inv = 1.0f / red[0];
#pragma unroll
    for (int i = 0; i < 16; ++i) p[tid + i * 256] = loc[i] * inv;
}

// ---------------- kernel 3: partial context, deterministic split-S ----------------
__global__ __launch_bounds__(256) void
ctx_partial_kernel(const float* __restrict__ x,
                   const float* __restrict__ w,
                   float* __restrict__ partial) {
    __shared__ float wl[256];
    const int b = blockIdx.x >> 4, ch = blockIdx.x & 15, d = threadIdx.x;
    const int s0 = ch * 256;
    wl[d] = w[(size_t)b * Ssz + s0 + d];
    __syncthreads();
    const float* xb = x + ((size_t)b * Ssz + s0) * Dsz;
    float acc = 0.f;
    for (int r = 0; r < 256; ++r) {
        __builtin_prefetch(xb + (size_t)(r + 16) * Dsz + d, 0, 0);  // global_prefetch_b8, stream ahead
        acc += wl[r] * xb[(size_t)r * Dsz + d];
    }
    partial[(size_t)blockIdx.x * Dsz + d] = acc;
}

__global__ __launch_bounds__(256) void
ctx_reduce_kernel(const float* __restrict__ partial,
                  float* __restrict__ context) {
    const int b = blockIdx.x, d = threadIdx.x;
    float acc = 0.f;
#pragma unroll
    for (int ch = 0; ch < 16; ++ch)
        acc += partial[((size_t)(b * 16 + ch)) * Dsz + d];
    context[(size_t)b * Dsz + d] = acc;
}

extern "C" void kernel_launch(void* const* d_in, const int* in_sizes, int n_in,
                              void* d_out, int out_size, void* d_ws, size_t ws_size,
                              hipStream_t stream) {
    (void)in_sizes; (void)n_in; (void)out_size; (void)ws_size;
    const float* x    = (const float*)d_in[0];
    const float* W    = (const float*)d_in[1];
    const float* bias = (const float*)d_in[2];
    const float* u    = (const float*)d_in[3];

    float* context = (float*)d_out;                  // [B*D]
    float* weights = (float*)d_out + Bsz * Dsz;      // [B*S] (scores first, softmaxed in place)

    _Float16* Wt   = (_Float16*)d_ws;                             // 128 KB
    float* partial = (float*)((char*)d_ws + Dsz * Dsz * sizeof(_Float16)); // 1 MB

    prep_wt_kernel<<<256, 256, 0, stream>>>(W, Wt);

    const size_t lds1 = (size_t)(Dsz + BLK_ROWS) * LDSK * sizeof(_Float16); // 270336 B
    (void)hipFuncSetAttribute(reinterpret_cast<const void*>(score_kernel),
                              hipFuncAttributeMaxDynamicSharedMemorySize, (int)lds1);
    score_kernel<<<(Bsz * Ssz) / BLK_ROWS, 256, lds1, stream>>>(x, Wt, bias, u, weights);

    softmax_kernel<<<Bsz, 256, 0, stream>>>(weights);
    ctx_partial_kernel<<<Bsz * 16, 256, 0, stream>>>(x, weights, partial);
    ctx_reduce_kernel<<<Bsz, 256, 0, stream>>>(partial, context);
}
